// BoundaryPredictor_8804682957284
// MI455X (gfx1250) — matrix operand order repaired and checked by reading, not compiled, not run on hardware
//
#include <hip/hip_runtime.h>
#include <hip/hip_bf16.h>
#include <stdint.h>

// ---------------------------------------------------------------------------
// BoundaryPredictor for MI455X (gfx1250, wave32).
// Bandwidth-bound (~86 MB total -> ~4us at 23.3 TB/s). Product-domain rewrite
// removes all logs (window terms >= 0.1, so the 21-term product >= 1e-21 >
// FLT_MIN and fp32 product is safe). CDNA5 paths:
//   - global_load_async_to_lds_b128 staging of pf/pb (contiguous block image)
//   - v_wmma_f32_16x16x32_f16 for the moment sums (S0,S1,S2), 16 elems/op
//   - global_store_async_from_lds_b128 for the dist output
// ---------------------------------------------------------------------------

typedef __attribute__((ext_vector_type(16))) _Float16 v16h;
typedef __attribute__((ext_vector_type(8)))  float    v8f;

#define F_LEN 21
#define D_LEN 41
#define S_LEN 61          // F_LEN + D_LEN - 1
#define TILE  64          // elements per workgroup (1 per lane, 2 waves)

__global__ __launch_bounds__(TILE) void boundary_predictor_kernel(
    const float* __restrict__ pf, const float* __restrict__ pb,
    const float* __restrict__ slop, const float* __restrict__ amp,
    float* __restrict__ out_dist, float* __restrict__ out_mean,
    float* __restrict__ out_ivar) {
  // LDS: exact images of the pf/pb blocks (stride 61 dwords, coprime with 64
  // banks -> conflict-free per-lane reads) + normalized dist (stride 41,
  // also coprime with 64, contiguous = global layout) + zero pad for WMMA.
  __shared__ __align__(16) float pf_s[TILE * S_LEN];
  __shared__ __align__(16) float pb_s[TILE * S_LEN];
  __shared__ __align__(16) float dist_s[TILE * D_LEN + 64];

  const int tid  = threadIdx.x;           // 0..63, one element per lane
  const int lane = tid & 31;
  const int wv   = tid >> 5;
  const int blk  = blockIdx.x * TILE;
  const int gid  = blk + tid;

  // ---- Stage pf/pb via async tensor path: 64*61*4 = 15616 B contiguous ----
  const uint64_t pf_src = (uint64_t)(uintptr_t)(pf + (size_t)blk * S_LEN);
  const uint64_t pb_src = (uint64_t)(uintptr_t)(pb + (size_t)blk * S_LEN);
  const uint32_t lds_pf = (uint32_t)(uintptr_t)pf_s;
  const uint32_t lds_pb = (uint32_t)(uintptr_t)pb_s;
  const int n128 = TILE * S_LEN * 4 / 16;  // 976 b128 transfers per array
  for (int i = tid; i < n128; i += TILE) {
    uint32_t off = (uint32_t)i * 16u;
    asm volatile("global_load_async_to_lds_b128 %0, %1, %2"
                 :: "v"(lds_pf + off), "v"(off), "s"(pf_src) : "memory");
    asm volatile("global_load_async_to_lds_b128 %0, %1, %2"
                 :: "v"(lds_pb + off), "v"(off), "s"(pb_src) : "memory");
  }
  asm volatile("s_wait_asynccnt 0" ::: "memory");
  __syncthreads();

  // ---- Per-lane scalars: lookup weights lf[f] and min_var ----
  const float sl = slop[gid];
  const float am = amp[gid];
  const float i2s = 1.0f / (2.0f * sl);
  float lf[F_LEN];
#pragma unroll
  for (int f = 0; f < F_LEN; ++f) {
    float x = (float)f - (float)(F_LEN - 1) * 0.5f;   // f - 10
    float y = x * i2s;
    float e2 = __expf(2.0f * y);                       // v_exp_f32 path
    float th = (e2 - 1.0f) * __builtin_amdgcn_rcpf(e2 + 1.0f);
    lf[f] = 0.5f - am * th;
  }
  const float z  = 2.0f * am;
  const float at = 0.5f * __logf((1.0f + z) * __builtin_amdgcn_rcpf(1.0f - z));
  const float minv = fmaxf(1.0f / (2.0f * at * at), sl);

  // ---- Sliding-window product (log-free; terms >= 0.1 so no underflow) ----
  const float* pfe = pf_s + tid * S_LEN;
  const float* pbe = pb_s + tid * S_LEN;
  float dist[D_LEN];
  float sum = 0.0f;
#pragma unroll
  for (int d = 0; d < D_LEN; ++d) {
    float p = 1.0f;
#pragma unroll
    for (int f = 0; f < F_LEN; ++f) {
      float a = pfe[d + f];                // CSE'd across d by the compiler
      float b = pbe[d + f];
      p *= fmaf(lf[f], a - b, b);          // t = pb + lf*(pf-pb)
    }
    dist[d] = p;
    sum += p;
  }

  // ---- L1 normalize, publish to LDS (stride 41 = global image) ----
  const float inv = 1.0f / fmaxf(sum, 1e-12f);
#pragma unroll
  for (int d = 0; d < D_LEN; ++d) dist_s[tid * D_LEN + d] = dist[d] * inv;
  dist_s[TILE * D_LEN + tid] = 0.0f;       // zero pad for WMMA k>=41 reads
  __syncthreads();

  // ---- Async store dist block (64*41*4 = 10496 B contiguous) ----
  const uint64_t dst = (uint64_t)(uintptr_t)(out_dist + (size_t)blk * D_LEN);
  const uint32_t lds_d = (uint32_t)(uintptr_t)dist_s;
  const int s128 = TILE * D_LEN * 4 / 16;  // 656 b128 transfers
  for (int i = tid; i < s128; i += TILE) {
    uint32_t off = (uint32_t)i * 16u;
    asm volatile("global_store_async_from_lds_b128 %0, %1, %2"
                 :: "v"(off), "v"(lds_d + off), "s"(dst) : "memory");
  }

  // ---- Moments via v_wmma_f32_16x16x32_f16: D[m,n] = sum_k A[m,k]*dist_n[k]
  // A rows: m=0 -> 1, m=1 -> k, m=2 -> k^2 (f16-exact: k<=40, k^2<=1600),
  // zero for k >= 41 and rows >= 3. Built BRANCHLESSLY as the polynomial
  //   g(k) * (w0 + k*w1 + k^2*w2),  w_m = per-lane row indicator.
  // B columns n = 16 elements of this wave; two chains cover all 32 lanes.
  const int hi   = lane >> 4;              // half-wave select
  const int mrow = lane & 15;
  const int col0 = wv * 32 + mrow;         // chain 0: elements base+0..15
  const int col1 = col0 + 16;              // chain 1: elements base+16..31
  const float w0  = (mrow == 0) ? 1.0f : 0.0f;   // v_cndmask, no branches
  const float w1  = (mrow == 1) ? 1.0f : 0.0f;
  const float w2  = (mrow == 2) ? 1.0f : 0.0f;
  const float hi8 = hi ? 8.0f : 0.0f;
  v8f c0 = {}, c1 = {};
#pragma unroll
  for (int chunk = 0; chunk < 2; ++chunk) {
    v16h A, B0, B1;
#pragma unroll
    for (int p = 0; p < 8; ++p) {
      // 16-bit A layout: lanes 0-15 hold K 0..7 / 16..23; lanes 16-31 +8
      int kbase = ((p < 4) ? (2 * p) : (16 + 2 * (p - 4))) + chunk * 32;
      float k0 = (float)kbase + hi8;
      float k1 = k0 + 1.0f;
      float g0 = (k0 < (float)D_LEN) ? 1.0f : 0.0f;
      float g1 = (k1 < (float)D_LEN) ? 1.0f : 0.0f;
      A[2 * p]     = (_Float16)(g0 * fmaf(k0, fmaf(k0, w2, w1), w0));
      A[2 * p + 1] = (_Float16)(g1 * fmaf(k1, fmaf(k1, w2, w1), w0));
      // 16-bit B layout (same as A): lanes 0-15 hold K 0..7 / 16..23; lanes 16-31 +8
      int kb = kbase + (hi ? 8 : 0);
      B0[2 * p]     = (_Float16)dist_s[col0 * D_LEN + kb];
      B0[2 * p + 1] = (_Float16)dist_s[col0 * D_LEN + kb + 1];
      B1[2 * p]     = (_Float16)dist_s[col1 * D_LEN + kb];
      B1[2 * p + 1] = (_Float16)dist_s[col1 * D_LEN + kb + 1];
    }
    c0 = __builtin_amdgcn_wmma_f32_16x16x32_f16(false, A, false, B0,
                                                (short)0, c0, false, false);
    c1 = __builtin_amdgcn_wmma_f32_16x16x32_f16(false, A, false, B1,
                                                (short)0, c1, false, false);
  }

  // D layout: VGPR r, lanes 0-15 = row r, col = lane. Lane L<16 owns chain0
  // col L (its own element); lane L>=16 needs chain1 col L-16 -> shfl_xor 16.
  float t0 = __shfl_xor(c1[0], 16, 32);
  float t1 = __shfl_xor(c1[1], 16, 32);
  float t2 = __shfl_xor(c1[2], 16, 32);
  float S0 = hi ? t0 : c0[0];
  float S1 = hi ? t1 : c0[1];
  float S2 = hi ? t2 : c0[2];

  float mt   = S1;                                       // mean_tmp
  float vt   = S2 - 2.0f * mt * S1 + mt * mt * S0;       // var_tmp
  float var  = fmaxf(vt, minv);
  out_mean[gid] = mt - (float)(D_LEN - 1) * 0.5f;        // mt - 20
  out_ivar[gid] = 1.0f / var;
}

extern "C" void kernel_launch(void* const* d_in, const int* in_sizes, int n_in,
                              void* d_out, int out_size, void* d_ws, size_t ws_size,
                              hipStream_t stream) {
  // inputs: [0] lines_feature (unused), [1] pf, [2] pb, [3] slop, [4] amp
  const float* pf   = (const float*)d_in[1];
  const float* pb   = (const float*)d_in[2];
  const float* slop = (const float*)d_in[3];
  const float* amp  = (const float*)d_in[4];
  const int total = in_sizes[3];                 // B*L = 131072
  float* out_dist = (float*)d_out;               // [total, 41]
  float* out_mean = out_dist + (size_t)total * D_LEN;
  float* out_ivar = out_mean + total;
  const int blocks = total / TILE;               // 2048 workgroups
  boundary_predictor_kernel<<<dim3(blocks), dim3(TILE), 0, stream>>>(
      pf, pb, slop, amp, out_dist, out_mean, out_ivar);
}
